// AttentionModel_83597243449670
// MI455X (gfx1250) — matrix operand order repaired and checked by reading, not compiled, not run on hardware
//
#include <hip/hip_runtime.h>
#include <hip/hip_bf16.h>

typedef __attribute__((ext_vector_type(8)))  _Float16 v8h;
typedef __attribute__((ext_vector_type(16))) _Float16 v16h;
typedef __attribute__((ext_vector_type(8)))  float    v8f;

#define DEV __device__ __forceinline__

DEV v16h cat16(v8h lo, v8h hi) {
    return __builtin_shufflevector(lo, hi, 0,1,2,3,4,5,6,7,8,9,10,11,12,13,14,15);
}
// A fragment (16x32 f16, row-major source, row = lane&15):
// elements 0..7 -> K = 8*half + 0..7 ; elements 8..15 -> K = 16 + 8*half + 0..7
DEV v16h load_a32(const _Float16* p /* row base + 8*half, 16B aligned */) {
    return cat16(*(const v8h*)p, *(const v8h*)(p + 16));
}
// Packed B fragment: 16 halves per lane (K = 8*half + e for e<8, 16 + 8*half + e-8 for e>=8, N = lane&15)
DEV v16h load_pk(const _Float16* p) { return *(const v16h*)p; }

DEV v8f wmma(v16h a, v16h b, v8f c) {
    return __builtin_amdgcn_wmma_f32_16x16x32_f16(false, a, false, b, (short)0, c, false, false);
}

DEV v8f zero8() { v8f z = {0.f,0.f,0.f,0.f,0.f,0.f,0.f,0.f}; return z; }

// ---------------------------------------------------------------- instnorm
__global__ void instnorm_kernel(const float* __restrict__ x,
                                float* __restrict__ xn32,
                                _Float16* __restrict__ xn16) {
    const int L = 32768;
    int b = blockIdx.x;
    const float* row = x + (size_t)b * L;
    float s = 0.f, s2 = 0.f;
    for (int i = threadIdx.x; i < L; i += blockDim.x) {
        float v = row[i]; s += v; s2 += v * v;
    }
    __shared__ float sh0[256], sh1[256];
    sh0[threadIdx.x] = s; sh1[threadIdx.x] = s2;
    __syncthreads();
    for (int off = 128; off > 0; off >>= 1) {
        if (threadIdx.x < off) {
            sh0[threadIdx.x] += sh0[threadIdx.x + off];
            sh1[threadIdx.x] += sh1[threadIdx.x + off];
        }
        __syncthreads();
    }
    float mean = sh0[0] / (float)L;
    float var  = (sh1[0] - (float)L * mean * mean) / (float)(L - 1);
    var = var < 0.f ? 0.f : var;
    float inv = 1.0f / (sqrtf(var) + 1e-5f);
    for (int i = threadIdx.x; i < L; i += blockDim.x) {
        float v = (row[i] - mean) * inv;
        xn32[(size_t)b * L + i] = v;
        xn16[(size_t)b * L + i] = (_Float16)v;
    }
}

// ------------------------------------------------- weight -> packed B frags
__global__ void pack_w_kernel(const float* __restrict__ W, int N,
                              int Ktiles, int Ntiles, _Float16* __restrict__ dst) {
    int gid  = blockIdx.x * blockDim.x + threadIdx.x;
    int lane = gid & 31;
    int f    = gid >> 5;
    if (f >= Ktiles * Ntiles) return;
    int kt = f / Ntiles, nt = f % Ntiles;
    int half = lane >> 4, nl = lane & 15;
    int n = nt * 16 + nl;
    _Float16* out = dst + ((size_t)f * 32 + lane) * 16;
    #pragma unroll
    for (int e = 0; e < 16; ++e) {
        int k = kt * 32 + half * 8 + (e < 8 ? e : e + 8);
        out[e] = (_Float16)W[(size_t)k * N + n];
    }
}

// --------------------------------------------- repack K^T and V into frags
__global__ void repack_kv_kernel(const _Float16* __restrict__ qkv,
                                 _Float16* __restrict__ Kp,
                                 _Float16* __restrict__ Vp) {
    int gid  = blockIdx.x * blockDim.x + threadIdx.x;
    int lane = gid & 31;
    int f    = gid >> 5;              // 0..16383
    int b    = f >> 9;                // 512 frags per batch for both K and V
    int rem  = f & 511;
    int half = lane >> 4, nl = lane & 15;
    if (blockIdx.y == 0) {
        // K^T as B matrix: B[k=d (256)][n=key (1024)] ; kt over 8 d-chunks, nt over 64 key tiles
        int kt = rem >> 6, nt = rem & 63;
        int key = nt * 16 + nl;
        const _Float16* src = qkv + ((size_t)(b * 1024 + key)) * 768 + 256 + kt * 32 + half * 8;
        _Float16* out = Kp + ((size_t)f * 32 + lane) * 16;
        #pragma unroll
        for (int e = 0; e < 16; ++e) out[e] = src[e < 8 ? e : e + 8];
    } else {
        // V as B matrix: B[k=key][n=d] ; kt over 32 key-chunks, nt over 16 d tiles
        int kt = rem >> 4, nt = rem & 15;
        int d = nt * 16 + nl;
        _Float16* out = Vp + ((size_t)f * 32 + lane) * 16;
        #pragma unroll
        for (int e = 0; e < 16; ++e) {
            int key = kt * 32 + half * 8 + (e < 8 ? e : e + 8);
            out[e] = qkv[((size_t)(b * 1024 + key)) * 768 + 512 + d];
        }
    }
}

// ------------------------------------------------------------ WMMA GEMM
// C[M,N] = A[M,K]_f16 @ Bpacked + bias ; one wave per 16-row tile.
template<int KT, int NT>
__global__ __launch_bounds__(128, 1)
void gemm_kernel(const _Float16* __restrict__ A, int lda,
                 const _Float16* __restrict__ Bp,
                 const float* __restrict__ bias,
                 _Float16* __restrict__ C, int ldc, int Mtiles) {
    int wave = threadIdx.x >> 5;
    int lane = threadIdx.x & 31;
    int mt = blockIdx.x * (blockDim.x >> 5) + wave;
    if (mt >= Mtiles) return;
    int half = lane >> 4, nl = lane & 15;

    v16h af[KT];
    const _Float16* arow = A + (size_t)(mt * 16 + nl) * lda;
    #pragma unroll
    for (int kt = 0; kt < KT; ++kt)
        af[kt] = load_a32(arow + kt * 32 + half * 8);

    for (int nt = 0; nt < NT; ++nt) {
        v8f acc = zero8();
        #pragma unroll
        for (int kt = 0; kt < KT; ++kt) {
            const _Float16* bp = Bp + (((size_t)kt * NT + nt) * 32 + lane) * 16;
            acc = wmma(af[kt], load_pk(bp), acc);
        }
        float bn = bias[nt * 16 + nl];
        #pragma unroll
        for (int r = 0; r < 8; ++r) {
            float v = acc[r] + bn;
            C[(size_t)(mt * 16 + r + half * 8) * ldc + nt * 16 + nl] = (_Float16)v;
        }
    }
}

// ------------------------------------------------------ flash attention
// One wave per 16-query tile. Q tile staged in LDS (keeps VGPR pressure
// at ~190 so the 128-VGPR O accumulator never spills to scratch).
__global__ __launch_bounds__(32, 1)
void attn_kernel(const _Float16* __restrict__ qkv,
                 const _Float16* __restrict__ Kp,
                 const _Float16* __restrict__ Vp,
                 _Float16* __restrict__ o) {
    int qt = blockIdx.x;     // 0..63 query tiles
    int b  = blockIdx.y;     // 0..31 batch
    int lane = threadIdx.x;
    int half = lane >> 4, nl = lane & 15;
    __shared__ alignas(16) _Float16 qtile[16 * 256];
    __shared__ alignas(16) _Float16 pbuf[16 * 32];

    // Cooperative, coalesced copy of the 16x256 Q tile into LDS.
    {
        const _Float16* qbase = qkv + ((size_t)(b * 1024 + qt * 16)) * 768;
        #pragma unroll
        for (int c = 0; c < 16; ++c) {
            int chunk = c * 32 + lane;          // 512 chunks of 8 halves
            int row = chunk >> 5;               // 32 chunks per row
            int off = (chunk & 31) * 8;
            *(v8h*)(qtile + row * 256 + off) = *(const v8h*)(qbase + (size_t)row * 768 + off);
        }
    }
    __syncthreads();

    v8f accv[16];
    #pragma unroll
    for (int i = 0; i < 16; ++i) accv[i] = zero8();
    float mrow[8], lrow[8];
    #pragma unroll
    for (int r = 0; r < 8; ++r) { mrow[r] = -3.0e38f; lrow[r] = 0.f; }

    const float scale = 0.0625f;  // 1/sqrt(256)
    const _Float16* qlds = qtile + nl * 256 + half * 8;   // A-frag base for this lane
    int nkb = (qt >> 1) + 1;
    for (int kb = 0; kb < nkb; ++kb) {
        v8f s0 = zero8(), s1 = zero8();
        const _Float16* kbase = Kp + (((size_t)b * 8 * 64 + kb * 2) * 32 + lane) * 16;
        #pragma unroll
        for (int kc = 0; kc < 8; ++kc) {
            v16h qf = load_a32(qlds + kc * 32);
            const _Float16* k0 = kbase + (size_t)kc * 64 * 32 * 16;
            s0 = wmma(qf, load_pk(k0), s0);
            s1 = wmma(qf, load_pk(k0 + 32 * 16), s1);
        }
        int key0 = kb * 32 + nl, key1 = key0 + 16;
        #pragma unroll
        for (int r = 0; r < 8; ++r) {
            int qi = qt * 16 + r + half * 8;
            float x0 = s0[r] * scale; if (key0 > qi) x0 = -1e30f;
            float x1 = s1[r] * scale; if (key1 > qi) x1 = -1e30f;
            float bm = fmaxf(x0, x1);
            bm = fmaxf(bm, __shfl_xor(bm, 1, 32));
            bm = fmaxf(bm, __shfl_xor(bm, 2, 32));
            bm = fmaxf(bm, __shfl_xor(bm, 4, 32));
            bm = fmaxf(bm, __shfl_xor(bm, 8, 32));
            float mnew = fmaxf(mrow[r], bm);
            float corr = __expf(mrow[r] - mnew);
            mrow[r] = mnew;
            float p0 = __expf(x0 - mnew);
            float p1 = __expf(x1 - mnew);
            float rs = p0 + p1;
            rs += __shfl_xor(rs, 1, 32);
            rs += __shfl_xor(rs, 2, 32);
            rs += __shfl_xor(rs, 4, 32);
            rs += __shfl_xor(rs, 8, 32);
            lrow[r] = lrow[r] * corr + rs;
            #pragma unroll
            for (int dt = 0; dt < 16; ++dt) accv[dt][r] *= corr;
            int row = r + half * 8;
            pbuf[row * 32 + nl]      = (_Float16)p0;
            pbuf[row * 32 + 16 + nl] = (_Float16)p1;
        }
        __syncthreads();
        v16h pf = load_a32(pbuf + nl * 32 + half * 8);
        const _Float16* vbase = Vp + ((((size_t)b * 32 + kb) * 16) * 32 + lane) * 16;
        #pragma unroll
        for (int dt = 0; dt < 16; ++dt)
            accv[dt] = wmma(pf, load_pk(vbase + (size_t)dt * 32 * 16), accv[dt]);
        __syncthreads();
    }
    #pragma unroll
    for (int r = 0; r < 8; ++r) {
        float inv = 1.0f / lrow[r];
        size_t rowoff = ((size_t)(b * 1024 + qt * 16 + r + half * 8)) * 256;
        #pragma unroll
        for (int dt = 0; dt < 16; ++dt)
            o[rowoff + dt * 16 + nl] = (_Float16)(accv[dt][r] * inv);
    }
}

// --------------------------------- fused out-proj + head + shifted MSE loss
__global__ __launch_bounds__(32, 1)
void head_loss_kernel(const _Float16* __restrict__ o,
                      const _Float16* __restrict__ Wpo,
                      const float* __restrict__ b_out,
                      const _Float16* __restrict__ Wph,
                      const float* __restrict__ b_head,
                      const float* __restrict__ xn32,
                      float* __restrict__ loss) {
    int mt = blockIdx.x;       // 0..2047 (16-row tiles over 32768 rows)
    int lane = threadIdx.x;
    int half = lane >> 4, nl = lane & 15;
    __shared__ alignas(16) _Float16 out2[16 * 256];

    v16h af[8];
    const _Float16* ob = o + ((size_t)(mt * 16 + nl)) * 256;
    #pragma unroll
    for (int kc = 0; kc < 8; ++kc)
        af[kc] = load_a32(ob + kc * 32 + half * 8);

    #pragma unroll
    for (int nt = 0; nt < 16; ++nt) {
        v8f acc = zero8();
        #pragma unroll
        for (int kt = 0; kt < 8; ++kt) {
            const _Float16* bp = Wpo + (((size_t)kt * 16 + nt) * 32 + lane) * 16;
            acc = wmma(af[kt], load_pk(bp), acc);
        }
        float bn = b_out[nt * 16 + nl];
        #pragma unroll
        for (int r = 0; r < 8; ++r)
            out2[(r + half * 8) * 256 + nt * 16 + nl] = (_Float16)(acc[r] + bn);
    }
    __syncthreads();

    v16h a2[8];
    #pragma unroll
    for (int kc = 0; kc < 8; ++kc)
        a2[kc] = load_a32(out2 + nl * 256 + kc * 32 + half * 8);

    float lsum = 0.f;
    #pragma unroll
    for (int nt = 0; nt < 2; ++nt) {
        v8f acc = zero8();
        #pragma unroll
        for (int kt = 0; kt < 8; ++kt) {
            const _Float16* bp = Wph + (((size_t)kt * 2 + nt) * 32 + lane) * 16;
            acc = wmma(a2[kt], load_pk(bp), acc);
        }
        float bn = b_head[nt * 16 + nl];
        #pragma unroll
        for (int r = 0; r < 8; ++r) {
            int g = mt * 16 + r + half * 8;   // global row (b*1024 + t)
            int t = g & 1023;
            if (t < 1023) {
                float target = xn32[((size_t)g + 1) * 32 + nt * 16 + nl];
                float d = (acc[r] + bn) - target;
                lsum += d * d;
            }
        }
    }
    lsum *= (1.0f / (32.0f * 1023.0f * 32.0f));
    lsum += __shfl_xor(lsum, 16, 32);
    lsum += __shfl_xor(lsum, 8, 32);
    lsum += __shfl_xor(lsum, 4, 32);
    lsum += __shfl_xor(lsum, 2, 32);
    lsum += __shfl_xor(lsum, 1, 32);
    if (lane == 0) atomicAdd(loss, lsum);
}

// ---------------------------------------------------------------- launcher
extern "C" void kernel_launch(void* const* d_in, const int* in_sizes, int n_in,
                              void* d_out, int out_size, void* d_ws, size_t ws_size,
                              hipStream_t stream) {
    (void)in_sizes; (void)n_in; (void)out_size; (void)ws_size;
    const float* x      = (const float*)d_in[0];
    const float* W_proj = (const float*)d_in[1];
    const float* b_proj = (const float*)d_in[2];
    const float* W_qkv  = (const float*)d_in[3];
    const float* b_qkv  = (const float*)d_in[4];
    const float* W_out  = (const float*)d_in[5];
    const float* b_out  = (const float*)d_in[6];
    const float* W_head = (const float*)d_in[7];
    const float* b_head = (const float*)d_in[8];

    const size_t BL = 32u * 32768u;       // 1,048,576 elements
    const size_t M  = 32768;              // B*T rows

    char* ws = (char*)d_ws;
    size_t off = 0;
    auto take = [&](size_t bytes) { char* p = ws + off; off += (bytes + 255) & ~(size_t)255; return p; };

    float*    xn32 = (float*)   take(BL * 4);
    _Float16* xn16 = (_Float16*)take(BL * 2);
    _Float16* h    = (_Float16*)take(M * 256 * 2);
    _Float16* qkvb = (_Float16*)take(M * 768 * 2);
    _Float16* Kp   = (_Float16*)take((size_t)16384 * 512 * 2);
    _Float16* Vp   = (_Float16*)take((size_t)16384 * 512 * 2);
    _Float16* ob   = (_Float16*)take(M * 256 * 2);
    _Float16* Wpp  = (_Float16*)take((size_t)16  * 512 * 2);  // W_proj packed (1x16 tiles)
    _Float16* Wpq  = (_Float16*)take((size_t)384 * 512 * 2);  // W_qkv packed (8x48)
    _Float16* Wpo  = (_Float16*)take((size_t)128 * 512 * 2);  // W_out packed (8x16)
    _Float16* Wph  = (_Float16*)take((size_t)16  * 512 * 2);  // W_head packed (8x2)

    hipMemsetAsync(d_out, 0, sizeof(float), stream);

    instnorm_kernel<<<32, 256, 0, stream>>>(x, xn32, xn16);

    pack_w_kernel<<<(16  * 32 + 127) / 128, 128, 0, stream>>>(W_proj, 256, 1, 16, Wpp);
    pack_w_kernel<<<(384 * 32 + 127) / 128, 128, 0, stream>>>(W_qkv,  768, 8, 48, Wpq);
    pack_w_kernel<<<(128 * 32 + 127) / 128, 128, 0, stream>>>(W_out,  256, 8, 16, Wpo);
    pack_w_kernel<<<(16  * 32 + 127) / 128, 128, 0, stream>>>(W_head,  32, 8,  2, Wph);

    // h = patches @ W_proj + b_proj   (M x 32 @ 32 x 256)
    gemm_kernel<1, 16><<<512, 128, 0, stream>>>(xn16, 32, Wpp, b_proj, h, 256, 2048);
    // qkv = h @ W_qkv + b_qkv         (M x 256 @ 256 x 768)
    gemm_kernel<8, 48><<<512, 128, 0, stream>>>(h, 256, Wpq, b_qkv, qkvb, 768, 2048);

    repack_kv_kernel<<<dim3(4096, 2), 128, 0, stream>>>(qkvb, Kp, Vp);

    attn_kernel<<<dim3(64, 32), 32, 0, stream>>>(qkvb, Kp, Vp, ob);

    head_loss_kernel<<<2048, 32, 0, stream>>>(ob, Wpo, b_out, Wph, b_head, xn32, (float*)d_out);
}
